// GraphAttentionLayer_35038343201091
// MI455X (gfx1250) — matrix operand order repaired and checked by reading, not compiled, not run on hardware
//
#include <hip/hip_runtime.h>
#include <hip/hip_bf16.h>
#include <math.h>

typedef __attribute__((ext_vector_type(16))) _Float16 v16h;
typedef __attribute__((ext_vector_type(8)))  _Float16 v8h;
typedef __attribute__((ext_vector_type(8)))  float    v8f;
typedef __attribute__((ext_vector_type(4)))  float    v4f;

#define GAT_ALPHA 0.2f
#define GAT_NEG_INF (-9.0e15f)

// ---------------------------------------------------------------------------
// Kernel 0: W (256x256 f32, row-major [fin][fout]) -> WT f16 [fout][fin]
// so that WMMA B-fragments read K-contiguous halves.
// ---------------------------------------------------------------------------
__global__ __launch_bounds__(256) void k_cvtW(const float* __restrict__ W,
                                              _Float16* __restrict__ WT) {
    int idx = blockIdx.x * 256 + threadIdx.x;   // 65536 threads
    int fo = idx >> 8, fi = idx & 255;
    WT[fo * 256 + fi] = (_Float16)W[fi * 256 + fo];
}

// ---------------------------------------------------------------------------
// Kernel 1: Wh = h @ W via v_wmma_f32_16x16x32_f16.
// One wave per 16x16 output tile. 4096 rows x 256 cols -> 4096 tiles.
// Writes Wh f32 (row-major, for s1/s2) and WhT f16 ([b][col][row], K-contig
// for the attention GEMM's B-fragments).
// ---------------------------------------------------------------------------
__global__ __launch_bounds__(256) void k_gemm_hW(const float* __restrict__ h,
                                                 const _Float16* __restrict__ WT,
                                                 float* __restrict__ Wh,
                                                 _Float16* __restrict__ WhT) {
    int wid  = (blockIdx.x * 256 + threadIdx.x) >> 5;  // global wave id, 0..4095
    int lane = threadIdx.x & 31;
    int rt = wid >> 4;            // row tile 0..255
    int ct = wid & 15;            // col tile 0..15
    int m  = lane & 15;
    int g  = lane >> 4;
    int row = rt * 16 + m;        // A row for this lane (flat over B*N)
    int col = ct * 16 + m;        // B/C column for this lane

    v8f acc = {};
    const float* arow = h + (size_t)row * 256;
    const _Float16* bcol = WT + (size_t)col * 256;

    for (int kt = 0; kt < 8; ++kt) {
        int k0 = kt * 32;
        // A fragment: runs K = k0+8g..+7 and k0+16+8g..+7 (contiguous f32)
        v4f a0 = *(const v4f*)(arow + k0 + 8 * g);
        v4f a1 = *(const v4f*)(arow + k0 + 8 * g + 4);
        v4f a2 = *(const v4f*)(arow + k0 + 16 + 8 * g);
        v4f a3 = *(const v4f*)(arow + k0 + 16 + 8 * g + 4);
        v16h A;
#pragma unroll
        for (int t = 0; t < 4; ++t) {
            A[t]      = (_Float16)a0[t];
            A[4 + t]  = (_Float16)a1[t];
            A[8 + t]  = (_Float16)a2[t];
            A[12 + t] = (_Float16)a3[t];
        }
        // B fragment: K = k0+8g..+7 and k0+16+8g..+7
        v8h q0 = *(const v8h*)(bcol + k0 + 8 * g);
        v8h q1 = *(const v8h*)(bcol + k0 + 16 + 8 * g);
        v16h Bf;
#pragma unroll
        for (int t = 0; t < 8; ++t) { Bf[t] = q0[t]; Bf[8 + t] = q1[t]; }

        acc = __builtin_amdgcn_wmma_f32_16x16x32_f16(
            false, A, false, Bf, (short)0, acc, false, false);
    }

    // C layout: lane holds (M = v + 8g, N = col) for v = 0..7
    int b    = (rt * 16) >> 10;              // batch
    int rloc = rt * 16 - b * 1024 + 8 * g;   // row-in-batch of first element
    v8h hv;
#pragma unroll
    for (int v = 0; v < 8; ++v) {
        Wh[(size_t)(rt * 16 + v + 8 * g) * 256 + col] = acc[v];
        hv[v] = (_Float16)acc[v];
    }
    *(v8h*)(WhT + (size_t)b * 262144 + (size_t)col * 1024 + rloc) = hv;
}

// ---------------------------------------------------------------------------
// Kernel 2: s1[r] = Wh[r,:].a1 ; s2[r] = Wh[r,:].a2  (one wave per row)
// ---------------------------------------------------------------------------
__global__ __launch_bounds__(256) void k_s1s2(const float* __restrict__ Wh,
                                              const float* __restrict__ a,
                                              float* __restrict__ s1,
                                              float* __restrict__ s2) {
    int wid  = (blockIdx.x * 256 + threadIdx.x) >> 5;  // 0..4095
    int lane = threadIdx.x & 31;
    const float* r = Wh + (size_t)wid * 256;
    float p1 = 0.f, p2 = 0.f;
#pragma unroll
    for (int c = lane; c < 256; c += 32) {
        float v = r[c];
        p1 += v * a[c];
        p2 += v * a[256 + c];
    }
#pragma unroll
    for (int msk = 16; msk > 0; msk >>= 1) {
        p1 += __shfl_xor(p1, msk, 32);
        p2 += __shfl_xor(p2, msk, 32);
    }
    if (lane == 0) { s1[wid] = p1; s2[wid] = p2; }
}

// ---------------------------------------------------------------------------
// Kernel 3: fused scores + softmax + att@Wh + ELU.
// Block = (batch b, 16-row tile). 256 threads = 8 waves. 96 KB LDS.
//   Phase A: masked f32 scores into 64KB LDS slab (16 x 1024).
//   Phase B: rowwise softmax (half-wave per row); writes NORMALIZED f16
//            probabilities once into a 32KB LDS slab -> Phase C needs no
//            conversion/scaling VALU at all.
//   Phase C: 8 waves x (16x32) output tiles; A-fragments are two
//            ds_load_b128 of f16, B-fragments two global_load_b128 of
//            K-contiguous WhT; 2 WMMAs per K-step; ELU on C-fragments.
// ---------------------------------------------------------------------------
__global__ __launch_bounds__(256) void k_attn(const float* __restrict__ mz,
                                              const int* __restrict__ adj,
                                              const float* __restrict__ s1g,
                                              const float* __restrict__ s2g,
                                              const float* __restrict__ w1,
                                              const float* __restrict__ b1,
                                              const float* __restrict__ w2,
                                              const float* __restrict__ b2,
                                              const _Float16* __restrict__ WhT,
                                              float* __restrict__ out) {
    __shared__ float    sc[16 * 1024];       // 64 KB f32 scores / exp
    __shared__ _Float16 p16[16 * 1024];      // 32 KB normalized f16 probs

    int b  = blockIdx.x >> 6;
    int i0 = (blockIdx.x & 63) << 4;         // row tile start within batch
    int tid = threadIdx.x;

    // MLP weights in registers (fully unrolled H=32)
    float rw1[32], rb1[32], rw2[32];
#pragma unroll
    for (int t = 0; t < 32; ++t) { rw1[t] = w1[t]; rb1[t] = b1[t]; rw2[t] = w2[t]; }
    float rb2 = b2[0];

    const float* mzb  = mz  + b * 1024;
    const float* s1b  = s1g + b * 1024;
    const float* s2b  = s2g + b * 1024;
    const int*   adjb = adj + (size_t)b * 1024 * 1024;

    // ---- Phase A: scores ----
    for (int idx = tid; idx < 16 * 1024; idx += 256) {
        int i = idx >> 10, j = idx & 1023;
        int gi = i0 + i;
        // prefetch next stripe of the adj stream (speculative, gfx1250
        // global_prefetch_b8)
        __builtin_prefetch(adjb + gi * 1024 + j + 2048, 0, 0);
        float d = fabsf(mzb[gi] - mzb[j]);
        if (gi == 0 || j == 0) d = 0.0f;
        float dt = rb2;
#pragma unroll
        for (int t = 0; t < 32; ++t)
            dt += fmaxf(fmaf(d, rw1[t], rb1[t]), 0.0f) * rw2[t];
        float s = s1b[gi] + s2b[j];
        s = (s > 0.0f) ? s : GAT_ALPHA * s;
        float ev = s + dt;
        sc[idx] = (adjb[gi * 1024 + j] > 0) ? ev : GAT_NEG_INF;
    }
    __syncthreads();

    // ---- Phase B: rowwise softmax (lanes 0-15 -> row 2w, 16-31 -> 2w+1) ----
    int w    = tid >> 5;
    int lane = tid & 31;
    int r    = 2 * w + (lane >> 4);
    int sl   = lane & 15;
    float*    row  = sc  + r * 1024;
    _Float16* prow = p16 + r * 1024;

    float mx = GAT_NEG_INF;
    for (int j = sl; j < 1024; j += 16) mx = fmaxf(mx, row[j]);
#pragma unroll
    for (int msk = 8; msk > 0; msk >>= 1) mx = fmaxf(mx, __shfl_xor(mx, msk, 16));

    float sum = 0.0f;
    for (int j = sl; j < 1024; j += 16) {
        float ev = expf(row[j] - mx);
        row[j] = ev;                          // each lane touches only its own j
        sum += ev;
    }
#pragma unroll
    for (int msk = 8; msk > 0; msk >>= 1) sum += __shfl_xor(sum, msk, 16);
    float inv = 1.0f / sum;                   // all 16 lanes of the half have it
    for (int j = sl; j < 1024; j += 16)
        prow[j] = (_Float16)(row[j] * inv);   // normalized f16 probs, written once
    __syncthreads();

    // ---- Phase C: O(16x256) = P(16x1024) @ Wh_b(1024x256), f16 WMMA ----
    int m = lane & 15, g = lane >> 4;
    int n0 = 32 * w + m;                        // wave covers cols 32w..32w+31
    int n1 = n0 + 16;
    const _Float16* bt = WhT + (size_t)b * 262144;
    const _Float16* pr = p16 + m * 1024;        // A-lane owns row m entirely
    v8f acc0 = {}, acc1 = {};

    for (int kt = 0; kt < 32; ++kt) {
        int k0 = kt * 32;
        // A fragment: two 16-B LDS loads, no conversion needed
        v8h alo = *(const v8h*)(pr + k0 + 16 * g);
        v8h ahi = *(const v8h*)(pr + k0 + 16 * g + 8);
        v16h A;
#pragma unroll
        for (int t = 0; t < 8; ++t) { A[t] = alo[t]; A[8 + t] = ahi[t]; }

        v8h q0 = *(const v8h*)(bt + (size_t)n0 * 1024 + k0 + 16 * g);
        v8h q1 = *(const v8h*)(bt + (size_t)n0 * 1024 + k0 + 16 * g + 8);
        v16h B0;
#pragma unroll
        for (int t = 0; t < 8; ++t) { B0[t] = q0[t]; B0[8 + t] = q1[t]; }
        acc0 = __builtin_amdgcn_wmma_f32_16x16x32_f16(
            false, A, false, B0, (short)0, acc0, false, false);

        v8h q2 = *(const v8h*)(bt + (size_t)n1 * 1024 + k0 + 16 * g);
        v8h q3 = *(const v8h*)(bt + (size_t)n1 * 1024 + k0 + 16 * g + 8);
        v16h B1;
#pragma unroll
        for (int t = 0; t < 8; ++t) { B1[t] = q2[t]; B1[8 + t] = q3[t]; }
        acc1 = __builtin_amdgcn_wmma_f32_16x16x32_f16(
            false, A, false, B1, (short)0, acc1, false, false);
    }

    // ELU + store (C layout: lane holds rows v+8g, col n0 / n1)
    float* ob = out + ((size_t)b * 1024 + i0) * 256;
#pragma unroll
    for (int v = 0; v < 8; ++v) {
        int rr = v + 8 * g;
        float x0 = acc0[v]; x0 = (x0 > 0.f) ? x0 : (expf(x0) - 1.0f);
        float x1 = acc1[v]; x1 = (x1 > 0.f) ? x1 : (expf(x1) - 1.0f);
        ob[(size_t)rr * 256 + n0] = x0;
        ob[(size_t)rr * 256 + n1] = x1;
    }
}

// ---------------------------------------------------------------------------
extern "C" void kernel_launch(void* const* d_in, const int* in_sizes, int n_in,
                              void* d_out, int out_size, void* d_ws, size_t ws_size,
                              hipStream_t stream) {
    const float* h   = (const float*)d_in[0];   // (4,1024,256)
    const float* mz  = (const float*)d_in[1];   // (4,1024,1)
    const float* W   = (const float*)d_in[2];   // (256,256)
    const float* a   = (const float*)d_in[3];   // (512,1)
    const float* w1  = (const float*)d_in[4];   // (1,32)
    const float* b1  = (const float*)d_in[5];   // (32,)
    const float* w2  = (const float*)d_in[6];   // (32,1)
    const float* b2  = (const float*)d_in[7];   // (1,)
    const int*   adj = (const int*)d_in[8];     // (4,1024,1024)
    float* out = (float*)d_out;

    char* ws = (char*)d_ws;
    _Float16* WT   = (_Float16*)(ws);                         // 128 KB
    _Float16* WhT  = (_Float16*)(ws + 131072);                // 2 MB
    float*    Wh   = (float*)(ws + 131072 + 2097152);         // 4 MB
    float*    s1   = (float*)(ws + 131072 + 2097152 + 4194304);
    float*    s2   = s1 + 4096;

    k_cvtW   <<<256, 256, 0, stream>>>(W, WT);
    k_gemm_hW<<<512, 256, 0, stream>>>(h, WT, Wh, WhT);
    k_s1s2   <<<512, 256, 0, stream>>>(Wh, a, s1, s2);
    k_attn   <<<256, 256, 0, stream>>>(mz, adj, s1, s2, w1, b1, w2, b2,
                                       WhT, out);
}
